// Attention_68564857914022
// MI455X (gfx1250) — matrix operand order repaired and checked by reading, not compiled, not run on hardware
//
#include <hip/hip_runtime.h>
#include <hip/hip_bf16.h>

typedef __attribute__((ext_vector_type(16))) _Float16 v16h;
typedef __attribute__((ext_vector_type(8)))  _Float16 v8h;
typedef __attribute__((ext_vector_type(8)))  float    v8f;
typedef __attribute__((ext_vector_type(4)))  unsigned int u32x4;
typedef __attribute__((ext_vector_type(8)))  int      i32x8;
typedef __attribute__((ext_vector_type(4)))  int      i32x4;

static constexpr int kB  = 4;
static constexpr int kS  = 2048;
static constexpr int kD  = 768;
static constexpr int kH  = 12;
static constexpr int kHD = 64;
static constexpr int kM  = kB * kS;  // 8192 rows (B*S)
// log2(10000)/32 : theta = pos * 2^(-i * kLog)
static constexpr float kLog = 0.41524101186092029f;

static __device__ __forceinline__ v8f wmma_f16(v16h a, v16h b, v8f c) {
  return __builtin_amdgcn_wmma_f32_16x16x32_f16(false, a, false, b,
                                                (short)0, c, false, false);
}

static __device__ __forceinline__ v16h cat8(v8h lo, v8h hi) {
  return __builtin_shufflevector(lo, hi, 0, 1, 2, 3, 4, 5, 6, 7, 8, 9, 10, 11,
                                 12, 13, 14, 15);
}

// ---------------------------------------------------------------------------
// TDM: DMA a 2-D f16 tile (tile1 rows x tile0 contiguous elems, row stride
// stride0 elems) from global memory into LDS at byte offset ldsOff.
// D# layout per CDNA5 ISA. 2-D => groups 2,3 = 0.
// ---------------------------------------------------------------------------
static __device__ __forceinline__ void tdm_load_2d_f16(
    const _Float16* gsrc, unsigned ldsOff, unsigned tdim0, unsigned tdim1,
    unsigned tile0, unsigned tile1, unsigned stride0) {
  const unsigned long long ga = (unsigned long long)(uintptr_t)gsrc;
  u32x4 g0;
  g0[0] = 1u;                                                  // count=1
  g0[1] = ldsOff;                                              // lds_addr
  g0[2] = (unsigned)ga;                                        // gaddr[31:0]
  g0[3] = (unsigned)((ga >> 32) & 0x01FFFFFFu) | 0x80000000u;  // [56:32]|type=2
  i32x8 g1;
  g1[0] = 0x00010000;                                     // data_size = 2 B
  g1[1] = (int)((tdim0 & 0xFFFFu) << 16);                 // tensor_dim0 lo16
  g1[2] = (int)((tdim0 >> 16) | ((tdim1 & 0xFFFFu) << 16));
  g1[3] = (int)((tdim1 >> 16) | (tile0 << 16));           // tile_dim0
  g1[4] = (int)(tile1 & 0xFFFFu);                         // tile_dim1 (dim2=0)
  g1[5] = (int)stride0;                                   // dim0 stride lo32
  g1[6] = 0;
  g1[7] = 0;
  const i32x4 z4 = {0, 0, 0, 0};
#if defined(__clang_major__) && __clang_major__ >= 23
  const i32x8 z8 = {0, 0, 0, 0, 0, 0, 0, 0};
  __builtin_amdgcn_tensor_load_to_lds(g0, g1, z4, z4, z8, 0);
#else
  __builtin_amdgcn_tensor_load_to_lds(g0, g1, z4, z4, 0);
#endif
}

static __device__ __forceinline__ unsigned lds_off(const void* p) {
  // LDS generic address: low 32 bits are the LDS byte offset (ISA 10.2).
  return (unsigned)(uintptr_t)p;
}

// ---------------------------------------------------------------------------
// Kernel 0: one-shot fp32 -> f16 conversion (x and the four weight matrices),
// so every GEMM hot loop streams f16 with no v_cvt in the inner loop.
// ---------------------------------------------------------------------------
__global__ __launch_bounds__(256) void cvt_f16_kernel(
    const float* __restrict__ in, _Float16* __restrict__ out, int n8) {
  const int i = blockIdx.x * 256 + threadIdx.x;
  if (i >= n8) return;
  const float4 f0 = ((const float4*)in)[i * 2];
  const float4 f1 = ((const float4*)in)[i * 2 + 1];
  v8h o;
  o[0] = (_Float16)f0.x; o[1] = (_Float16)f0.y;
  o[2] = (_Float16)f0.z; o[3] = (_Float16)f0.w;
  o[4] = (_Float16)f1.x; o[5] = (_Float16)f1.y;
  o[6] = (_Float16)f1.z; o[7] = (_Float16)f1.w;
  *(v8h*)(out + (size_t)i * 8) = o;
}

// ---------------------------------------------------------------------------
// Kernel 1: fused QKV projection + bias + RoPE. Each wave owns a 16(seq)x64
// strip == one head, so the RoPE pair (d,d+32) is in-lane across acc tiles.
// The 64-row W strip (shared by all 8 waves) is TDM-staged into LDS,
// double-buffered over the 24 k-steps; next A fragment prefetched one step
// ahead. q pre-scaled by 1/sqrt(64).
// q,k -> [B,H,S,64] f16; v -> [B,H,64,S] f16 (transposed for attention).
// grid: x = 36 strips (q/k/v x 12 heads), y = 64 row blocks of 128
// ---------------------------------------------------------------------------
__global__ __launch_bounds__(256) void qkv_rope_kernel(
    const _Float16* __restrict__ xh, const int* __restrict__ pos_ids,
    const _Float16* __restrict__ wqh, const float* __restrict__ bq,
    const _Float16* __restrict__ wkh, const float* __restrict__ bk,
    const _Float16* __restrict__ wvh, const float* __restrict__ bv,
    _Float16* __restrict__ qf, _Float16* __restrict__ kf,
    _Float16* __restrict__ vT) {
  __shared__ _Float16 smW[2 * 64 * 32];  // 2 bufs x (64 rows x 32 k) = 8 KB

  const int lane = threadIdx.x & 31;
  const int wave = threadIdx.x >> 5;
  const int l16  = lane & 15;
  const bool hi  = lane >= 16;

  const int strip   = blockIdx.x;  // 0..35
  const int mat     = strip / kH;  // 0=q 1=k 2=v
  const int head    = strip % kH;
  const int rowBase = blockIdx.y * 128 + wave * 16;

  const _Float16* Wh  = (mat == 0) ? wqh : (mat == 1) ? wkh : wvh;
  const float* bias   = (mat == 0) ? bq : (mat == 1) ? bk : bv;
  const _Float16* Wstrip = Wh + (size_t)(head * 64) * kD;

  const int aOff = hi ? 8 : 0;   // A half-lane K offset
  const int bOff = hi ? 8 : 0;   // B half-lane K offset

  const _Float16* arow = xh + (size_t)(rowBase + l16) * kD;

  v8f c[4] = {v8f{}, v8f{}, v8f{}, v8f{}};

  if (wave == 0)  // preload k-step 0: 64 rows x 32 k
    tdm_load_2d_f16(Wstrip, lds_off(smW), 32, 64, 32, 64, kD);

  v16h aCur = cat8(*(const v8h*)(arow + aOff), *(const v8h*)(arow + aOff + 16));

#pragma unroll 1
  for (int k0 = 0; k0 < kD; k0 += 32) {
    const int buf = (k0 >> 5) & 1;
    __builtin_amdgcn_s_wait_tensorcnt(0);
    __syncthreads();
    if (wave == 0 && k0 + 32 < kD)
      tdm_load_2d_f16(Wstrip + (k0 + 32), lds_off(smW + (buf ^ 1) * 2048), 32,
                      64, 32, 64, kD);

    // prefetch next A fragment (global): latency spans this iteration
    const int kn = (k0 + 32 < kD) ? (k0 + 32) : k0;
    const v16h aNxt = cat8(*(const v8h*)(arow + kn + aOff),
                           *(const v8h*)(arow + kn + aOff + 16));

    // burst-load all four B fragments, then run the WMMAs back-to-back
    const _Float16* smb = smW + buf * 2048;
    v16h bfr[4];
#pragma unroll
    for (int t = 0; t < 4; ++t)
      bfr[t] = cat8(*(const v8h*)(smb + (t * 16 + l16) * 32 + bOff),
                    *(const v8h*)(smb + (t * 16 + l16) * 32 + bOff + 16));
#pragma unroll
    for (int t = 0; t < 4; ++t) c[t] = wmma_f16(aCur, bfr[t], c[t]);

    aCur = aNxt;
  }

  // bias
#pragma unroll
  for (int t = 0; t < 4; ++t) {
    const float bval = bias[head * 64 + t * 16 + l16];
#pragma unroll
    for (int r = 0; r < 8; ++r) c[t][r] += bval;
  }

  if (mat < 2) {  // RoPE on q,k; fold 1/sqrt(HD) into q
    const float inv0   = exp2f(-(float)l16 * kLog);
    const float inv1   = exp2f(-(float)(16 + l16) * kLog);
    const float qscale = (mat == 0) ? 0.125f : 1.0f;
#pragma unroll
    for (int r = 0; r < 8; ++r) {
      const int   row = rowBase + r + (hi ? 8 : 0);
      const float p   = (float)pos_ids[row];
      float s0, c0, s1, c1;
      __sincosf(p * inv0, &s0, &c0);
      __sincosf(p * inv1, &s1, &c1);
      const float x0 = c[0][r], x1 = c[1][r], x2 = c[2][r], x3 = c[3][r];
      c[0][r] = (x0 * c0 - x2 * s0) * qscale;
      c[2][r] = (x2 * c0 + x0 * s0) * qscale;
      c[1][r] = (x1 * c1 - x3 * s1) * qscale;
      c[3][r] = (x3 * c1 + x1 * s1) * qscale;
    }
  }

#pragma unroll
  for (int r = 0; r < 8; ++r) {
    const int row = rowBase + r + (hi ? 8 : 0);
    const int b   = row / kS;
    const int s   = row % kS;
    const size_t bh = (size_t)(b * kH + head);
#pragma unroll
    for (int t = 0; t < 4; ++t) {
      const int d = t * 16 + l16;
      const _Float16 v = (_Float16)c[t][r];
      if (mat == 0)      qf[(bh * kS + s) * kHD + d] = v;
      else if (mat == 1) kf[(bh * kS + s) * kHD + d] = v;
      else               vT[(bh * kHD + d) * kS + s] = v;
    }
  }
}

// ---------------------------------------------------------------------------
// Kernel 2: flash attention. All 8 waves of a block share (b,h); K/V 32-key
// chunks are TDM-staged into LDS once per block (double-buffered) instead of
// 8x-redundant global loads. Scores computed transposed (S^T = K @ Q^T) so
// exp(P^T) repacks into the A-layout of P@V purely in-lane. Online softmax;
// V B-fragment DS loads are issued before the softmax math to hide latency.
// grid: x = 16 (q blocks of 128), y = 48 (b*h)
// ---------------------------------------------------------------------------
__global__ __launch_bounds__(256) void flash_attn_kernel(
    const _Float16* __restrict__ qf, const _Float16* __restrict__ kf,
    const _Float16* __restrict__ vT, _Float16* __restrict__ attn) {
  // per buffer: K tile 32key x 64d (2048 h) then V tile 64d x 32key (2048 h)
  __shared__ _Float16 smem[2 * 4096];  // 16 KB

  const int lane = threadIdx.x & 31;
  const int wave = threadIdx.x >> 5;
  const int l16  = lane & 15;
  const bool hi  = lane >= 16;

  const int bh    = blockIdx.y;
  const int b     = bh / kH;
  const int h     = bh % kH;
  const int qBase = blockIdx.x * 128 + wave * 16;

  const _Float16* kHead = kf + (size_t)bh * kS * kHD;
  const _Float16* vHead = vT + (size_t)bh * kHD * kS;

  // Q B-fragments (reused across all key tiles): dims 0..31 and 32..63
  const _Float16* qrow =
      qf + (size_t)bh * kS * kHD + (size_t)(qBase + l16) * kHD;
  const int oB = hi ? 8 : 0;
  const v16h qb0 = cat8(*(const v8h*)(qrow + oB), *(const v8h*)(qrow + oB + 16));
  const v16h qb1 = cat8(*(const v8h*)(qrow + 32 + oB),
                        *(const v8h*)(qrow + 32 + oB + 16));

  v8f   o[4] = {v8f{}, v8f{}, v8f{}, v8f{}};
  float mrow = -INFINITY;
  float lrow = 0.0f;

  const int aOff = hi ? 8 : 0;   // A half-lane K offset
  const int vOff = hi ? 8 : 0;   // V B-fragment half-lane key offset

  if (wave == 0) {  // preload chunk 0
    tdm_load_2d_f16(kHead, lds_off(smem), 64, 32, 64, 32, kHD);
    tdm_load_2d_f16(vHead, lds_off(smem + 2048), 32, 64, 32, 64, kS);
  }

#pragma unroll 1
  for (int kc = 0; kc < kS; kc += 32) {
    const int buf = (kc >> 5) & 1;
    __builtin_amdgcn_s_wait_tensorcnt(0);
    __syncthreads();
    if (wave == 0 && kc + 32 < kS) {
      const unsigned nb = lds_off(smem + (buf ^ 1) * 4096);
      tdm_load_2d_f16(kHead + (size_t)(kc + 32) * kHD, nb, 64, 32, 64, 32,
                      kHD);
      tdm_load_2d_f16(vHead + (kc + 32), nb + 4096, 32, 64, 32, 64, kS);
    }
    const _Float16* smK = smem + buf * 4096;
    const _Float16* smV = smK + 2048;

    // ---- burst-load all K A-fragments, then both score-tile WMMA chains ----
    v16h akf[4];  // [sub][kstep]
#pragma unroll
    for (int sub = 0; sub < 2; ++sub) {
      const _Float16* kr = smK + (sub * 16 + l16) * kHD;
      akf[sub * 2 + 0] = cat8(*(const v8h*)(kr + aOff),
                              *(const v8h*)(kr + aOff + 16));
      akf[sub * 2 + 1] = cat8(*(const v8h*)(kr + aOff + 32),
                              *(const v8h*)(kr + aOff + 48));
    }
    v8f st[2];
#pragma unroll
    for (int sub = 0; sub < 2; ++sub) {
      v8f cc = {};
      cc = wmma_f16(akf[sub * 2 + 0], qb0, cc);
      cc = wmma_f16(akf[sub * 2 + 1], qb1, cc);
      st[sub] = cc;
    }

    // ---- issue V B-fragment loads early; softmax math hides DS latency ----
    v16h bvf[4];
#pragma unroll
    for (int t = 0; t < 4; ++t)
      bvf[t] = cat8(*(const v8h*)(smV + (t * 16 + l16) * 32 + vOff),
                    *(const v8h*)(smV + (t * 16 + l16) * 32 + vOff + 16));

    // ---- online softmax over this 32-key chunk (row q = l16) ----
    float cmax = st[0][0];
#pragma unroll
    for (int r = 0; r < 8; ++r) {
      cmax = fmaxf(cmax, st[0][r]);
      cmax = fmaxf(cmax, st[1][r]);
    }
    cmax = fmaxf(cmax, __shfl_xor(cmax, 16));
    const float mnew  = fmaxf(mrow, cmax);
    const float alpha = __expf(mrow - mnew);
    mrow = mnew;

    v16h  ap;  // A-layout fragment of P (in-lane repack)
    float ssum = 0.0f;
#pragma unroll
    for (int r = 0; r < 8; ++r) {
      const float e0 = __expf(st[0][r] - mnew);
      const float e1 = __expf(st[1][r] - mnew);
      ssum += e0 + e1;
      ap[r]     = (_Float16)e0;
      ap[8 + r] = (_Float16)e1;
    }
    ssum += __shfl_xor(ssum, 16);
    lrow = lrow * alpha + ssum;

    // ---- rescale O by alpha(q), q = r + (hi?8:0) ----
#pragma unroll
    for (int r = 0; r < 8; ++r) {
      const float ar = __shfl(alpha, hi ? (r + 8) : r, 32);
      o[0][r] *= ar; o[1][r] *= ar; o[2][r] *= ar; o[3][r] *= ar;
    }

    // ---- O += P @ V ----
#pragma unroll
    for (int t = 0; t < 4; ++t) o[t] = wmma_f16(ap, bvf[t], o[t]);
  }

  // ---- normalize by row sum and store f16 [B*S, D] ----
#pragma unroll
  for (int r = 0; r < 8; ++r) {
    const float lr  = __shfl(lrow, hi ? (r + 8) : r, 32);
    const float inv = 1.0f / lr;
    const size_t row = (size_t)(b * kS + qBase + r + (hi ? 8 : 0));
#pragma unroll
    for (int t = 0; t < 4; ++t)
      attn[row * kD + h * 64 + t * 16 + l16] = (_Float16)(o[t][r] * inv);
  }
}

// ---------------------------------------------------------------------------
// Kernel 3: output projection  out = attn_f16 @ Wo^T (fp32 out).
// Wo 64-row strip TDM-staged into LDS like kernel 1; A prefetched one step.
// grid: x = 12 (64-col strips), y = 64 (128-row blocks)
// ---------------------------------------------------------------------------
__global__ __launch_bounds__(256) void out_proj_kernel(
    const _Float16* __restrict__ attn, const _Float16* __restrict__ woh,
    float* __restrict__ out) {
  __shared__ _Float16 smW[2 * 64 * 32];  // 8 KB

  const int lane = threadIdx.x & 31;
  const int wave = threadIdx.x >> 5;
  const int l16  = lane & 15;
  const bool hi  = lane >= 16;

  const int nBase   = blockIdx.x * 64;
  const int rowBase = blockIdx.y * 128 + wave * 16;

  const int aOff = hi ? 8 : 0;
  const int bOff = hi ? 8 : 0;

  const _Float16* Wstrip = woh + (size_t)nBase * kD;
  const _Float16* arow   = attn + (size_t)(rowBase + l16) * kD;
  v8f c[4] = {v8f{}, v8f{}, v8f{}, v8f{}};

  if (wave == 0)
    tdm_load_2d_f16(Wstrip, lds_off(smW), 32, 64, 32, 64, kD);

  v16h aCur = cat8(*(const v8h*)(arow + aOff), *(const v8h*)(arow + aOff + 16));

#pragma unroll 1
  for (int k0 = 0; k0 < kD; k0 += 32) {
    const int buf = (k0 >> 5) & 1;
    __builtin_amdgcn_s_wait_tensorcnt(0);
    __syncthreads();
    if (wave == 0 && k0 + 32 < kD)
      tdm_load_2d_f16(Wstrip + (k0 + 32), lds_off(smW + (buf ^ 1) * 2048), 32,
                      64, 32, 64, kD);

    const int kn = (k0 + 32 < kD) ? (k0 + 32) : k0;
    const v16h aNxt = cat8(*(const v8h*)(arow + kn + aOff),
                           *(const v8h*)(arow + kn + aOff + 16));

    const _Float16* smb = smW + buf * 2048;
    v16h bfr[4];
#pragma unroll
    for (int t = 0; t < 4; ++t)
      bfr[t] = cat8(*(const v8h*)(smb + (t * 16 + l16) * 32 + bOff),
                    *(const v8h*)(smb + (t * 16 + l16) * 32 + bOff + 16));
#pragma unroll
    for (int t = 0; t < 4; ++t) c[t] = wmma_f16(aCur, bfr[t], c[t]);

    aCur = aNxt;
  }

#pragma unroll
  for (int r = 0; r < 8; ++r) {
    const size_t row = (size_t)(rowBase + r + (hi ? 8 : 0));
#pragma unroll
    for (int t = 0; t < 4; ++t)
      out[row * kD + nBase + t * 16 + l16] = c[t][r];
  }
}

// ---------------------------------------------------------------------------
extern "C" void kernel_launch(void* const* d_in, const int* in_sizes, int n_in,
                              void* d_out, int out_size, void* d_ws,
                              size_t ws_size, hipStream_t stream) {
  (void)in_sizes; (void)n_in; (void)out_size; (void)ws_size;
  const float* x   = (const float*)d_in[0];
  const int*   pos = (const int*)d_in[1];
  const float* wq  = (const float*)d_in[2];
  const float* bq  = (const float*)d_in[3];
  const float* wk  = (const float*)d_in[4];
  const float* bk  = (const float*)d_in[5];
  const float* wv  = (const float*)d_in[6];
  const float* bv  = (const float*)d_in[7];
  const float* wo  = (const float*)d_in[8];
  float* out = (float*)d_out;

  _Float16* ws = (_Float16*)d_ws;
  const size_t per = (size_t)kB * kH * kS * kHD;  // 6,291,456 (== kM*kD)
  const size_t wsz = (size_t)kD * kD;             // 589,824
  _Float16* qf  = ws;
  _Float16* kfp = ws + per;
  _Float16* vTp = ws + 2 * per;
  _Float16* at  = ws + 3 * per;  // f16 attention output [B*S, D]
  _Float16* xh  = ws + 4 * per;  // f16 hidden states   [B*S, D]
  _Float16* wqh = ws + 5 * per;
  _Float16* wkh = wqh + wsz;
  _Float16* wvh = wkh + wsz;
  _Float16* woh = wvh + wsz;

  // one-shot fp32 -> f16
  cvt_f16_kernel<<<(int)(per / 8 / 256), 256, 0, stream>>>(x, xh,
                                                           (int)(per / 8));
  cvt_f16_kernel<<<(int)(wsz / 8 / 256), 256, 0, stream>>>(wq, wqh,
                                                           (int)(wsz / 8));
  cvt_f16_kernel<<<(int)(wsz / 8 / 256), 256, 0, stream>>>(wk, wkh,
                                                           (int)(wsz / 8));
  cvt_f16_kernel<<<(int)(wsz / 8 / 256), 256, 0, stream>>>(wv, wvh,
                                                           (int)(wsz / 8));
  cvt_f16_kernel<<<(int)(wsz / 8 / 256), 256, 0, stream>>>(wo, woh,
                                                           (int)(wsz / 8));

  qkv_rope_kernel<<<dim3(36, kM / 128), 256, 0, stream>>>(
      xh, pos, wqh, bq, wkh, bk, wvh, bv, qf, kfp, vTp);
  flash_attn_kernel<<<dim3(kS / 128, kB * kH), 256, 0, stream>>>(qf, kfp, vTp,
                                                                 at);
  out_proj_kernel<<<dim3(kD / 64, kM / 128), 256, 0, stream>>>(at, woh, out);
}
